// FusionVADModel_9852654977416
// MI455X (gfx1250) — matrix operand order repaired and checked by reading, not compiled, not run on hardware
//
#include <hip/hip_runtime.h>

// ============================================================================
// FusionVADModel for MI455X (gfx1250, wave32, WMMA), round 3.
//
// seq_len==1 => softmax over one key == 1 => attention context == v-proj;
// q/k projections and ln1 are dead code.
//
// Round-3 change vs round 2: control code bloat / register pressure.
// GEMM loops stay fully unrolled (WMMA + fragment-load scheduling); the
// elementwise passes (LN output incl. erff-GELU, proj3, combined, staging)
// are forced rolled (#pragma unroll 1) so erff is expanded once per site,
// not 16x. Dynamic work unchanged; static VALU/SALU and spills shrink.
//
// Input flattening assumption: jax.tree-style (dict keys sorted at every
// level): audio_vad, params.(a2t_attn, aro, audio_proj, dom, gate, integ,
// t2a_attn, text_proj, val), text_vad; linear {b,w}, ln {b,g}.
// ============================================================================

typedef __attribute__((ext_vector_type(16))) __bf16        v16bf;
typedef __attribute__((ext_vector_type(8)))  float         v8f;
typedef __attribute__((ext_vector_type(8)))  unsigned int  v8u;
typedef __attribute__((ext_vector_type(4)))  unsigned int  v4u;

#define WAVES 4
// Wave-level LDS fence: all prior ds writes visible before subsequent reads.
#define WAVE_SYNC() asm volatile("s_wait_dscnt 0x0" ::: "memory")

// ---------- fp32 -> bf16 (round-to-nearest-even) ----------
__device__ __forceinline__ unsigned f2bf(float f) {
  unsigned u = __builtin_bit_cast(unsigned, f);
  return (u + 0x7FFFu + ((u >> 16) & 1u)) >> 16;
}

// ---------- workspace layout (dword offsets of bf16 B-fragment arrays) ----
// W[N][K] stored as [nt][kt][lane][8 dwords]:
//   lane L: column n = nt*16 + (L&15), k = kt*32 + (L>>4)*8 + (j&7) + (j>>3)*16, j = 0..15
enum : int {
  WS_A2T_V = 0,       // 128x128 : NT=8 KT=4  ->  8192 dwords
  WS_A2T_O = 8192,
  WS_T2A_V = 16384,
  WS_T2A_O = 24576,
  WS_INTEG = 32768,   // 128x256 : NT=8 KT=8  -> 16384 dwords
  WS_VAL   = 49152,   //  64x128 : NT=4 KT=4  ->  4096 dwords
  WS_ARO   = 53248,
  WS_DOM   = 57344,
  WS_GATE  = 61440,   //   2x512 pad N->16 : NT=1 KT=16 -> 4096 dwords
  WS_VFC2  = 65536,   //   1x64  pad N->16 : NT=1 KT=2  ->  512 dwords
  WS_AFC2  = 66048,
  WS_DFC2  = 66560,   // total 67072 dwords = 262 KB of d_ws
};

// ---------- prep: fp32 weight -> bf16 B-fragment swizzle (cols >= Nreal = 0)
__global__ void vad_prep_frag(const float* __restrict__ W,
                              unsigned int* __restrict__ dst,
                              int NT, int KT, int Nreal) {
  const int total = NT * KT * 256;           // dwords
  const int Ktot  = KT * 32;
  for (int t = blockIdx.x * blockDim.x + threadIdx.x; t < total;
       t += gridDim.x * blockDim.x) {
    int w  = t & 7;
    int L  = (t >> 3) & 31;
    int kt = (t >> 8) % KT;
    int nt = t / (KT * 256);
    int n  = nt * 16 + (L & 15);
    int kb = kt * 32 + ((L >> 4) * 8) + 2 * (w & 3) + (w >> 2) * 16;
    unsigned lo = 0, hi = 0;
    if (n < Nreal) {
      lo = f2bf(W[n * Ktot + kb + 0]);
      hi = f2bf(W[n * Ktot + kb + 1]);
    }
    dst[t] = lo | (hi << 16);
  }
}

// ---------- per-wave LDS working set ----------
struct WaveMem {
  float af[16 * 128];             // fp32 audio features
  float tf[16 * 128];             // fp32 text features
  float aat[16 * 128];            // fp32 a2t attention out
  float tat[16 * 128];            // fp32 t2a attention out (also proj-w scratch)
  float tmp[16 * 128];            // fp32 scratch (pre-LN GEMM outputs)
  unsigned short abf[16 * 128];   // bf16 af   (later: integ)
  unsigned short tbf[16 * 128];   // bf16 tf   (later: head hidden, ld 64)
  unsigned short aabf[16 * 128];  // bf16 aat
  unsigned short tabf[16 * 128];  // bf16 tat
  unsigned short cbf[16 * 256];   // bf16 kv|v scratch, then combined (ld 256)
  float xa[16 * 4];               // audio_vad rows (padded)
  float xt[16 * 4];               // text_vad rows
  float gts[16 * 2];              // per-row softmax gates
  float gsc[16 * 2];              // gate score staging
};
static_assert(sizeof(WaveMem) % 32 == 0, "WaveMem must stay 32B aligned");

struct Params {
  const float *audio, *text;
  const float *ap_w, *ap_b, *ap_lng, *ap_lnb;
  const float *tp_w, *tp_b, *tp_lng, *tp_lnb;
  const float *a2t_ln2g, *a2t_ln2b, *a2t_vb, *a2t_ob;
  const float *t2a_ln2g, *t2a_ln2b, *t2a_vb, *t2a_ob;
  const float *gate_b;
  const float *integ_b, *integ_lng, *integ_lnb;
  const float *h_fc1b[3], *h_lng[3], *h_lnb[3], *h_fc2b[3];
  const unsigned int* ws;
  float* out;
  int B;
};

// ---------- A fragment: two contiguous 16B runs of a row-major bf16 matrix -
__device__ __forceinline__ v16bf load_afrag(const unsigned short* base,
                                            int ldA, int kt, int lane) {
  const int m  = lane & 15;
  const int kb = (lane >> 4) * 8;
  const unsigned short* p = base + m * ldA + kt * 32 + kb;
  v4u lo = *(const v4u*)p;          // k = kb .. kb+7
  v4u hi = *(const v4u*)(p + 16);   // k = kb+16 .. kb+23
  v8u a;
  a[0] = lo[0]; a[1] = lo[1]; a[2] = lo[2]; a[3] = lo[3];
  a[4] = hi[0]; a[5] = hi[1]; a[6] = hi[2]; a[7] = hi[3];
  return __builtin_bit_cast(v16bf, a);
}

__device__ __forceinline__ v16bf load_bfrag(const unsigned int* __restrict__ f,
                                            int idx, int lane) {
  return __builtin_bit_cast(v16bf, *(const v8u*)&f[(idx * 32 + lane) * 8]);
}

// ---------- WMMA GEMM with fp32 and/or bf16 natural outputs ----------------
template <int KT, int NT, bool OUTF, bool OUTB, bool RES>
__device__ __forceinline__ void gemm(const unsigned short* A, int ldA,
                                     const unsigned int* __restrict__ bfrag,
                                     const float* __restrict__ bias,
                                     float* outf, unsigned short* outb,
                                     int ldw, const float* res, int lane) {
  const int col = lane & 15;
  const int mb  = (lane >> 4) * 8;
#pragma unroll
  for (int nt = 0; nt < NT; ++nt) {
    if (nt + 1 < NT)
      __builtin_prefetch(&bfrag[((nt + 1) * KT * 32 + lane) * 8], 0, 1);
    v8f acc = {0.f, 0.f, 0.f, 0.f, 0.f, 0.f, 0.f, 0.f};
#pragma unroll
    for (int kt = 0; kt < KT; ++kt) {
      v16bf a = load_afrag(A, ldA, kt, lane);
      v16bf b = load_bfrag(bfrag, nt * KT + kt, lane);
      acc = __builtin_amdgcn_wmma_f32_16x16x32_bf16(
          false, a, false, b, (short)0, acc, false, false);
    }
    const int n  = nt * 16 + col;
    const float bs = bias[n];
#pragma unroll
    for (int r = 0; r < 8; ++r) {      // D layout: lane L, reg r -> (mb+r, col)
      const int m = mb + r;
      float v = acc[r] + bs;
      if (RES)  v += res[m * ldw + n];
      if (OUTF) outf[m * ldw + n] = v;
      if (OUTB) outb[m * ldw + n] = (unsigned short)f2bf(v);
    }
  }
}

// ---------- LayerNorm (+opt GELU) -> fp32 and/or packed bf16 ---------------
// Output pass is kept rolled: one erff expansion per call site, not W/8.
template <int W, bool OUTF, bool OUTB, bool GELU>
__device__ __forceinline__ void ln16(const float* src, float* dstf,
                                     unsigned short* dstb, int ldb,
                                     const float* __restrict__ gp,
                                     const float* __restrict__ bp, int lane) {
  const int m = lane >> 1, part = lane & 1, half = W >> 1, base = part * half;
  const float4* s4 = (const float4*)(src + m * W + base);
  float sum = 0.f, ss = 0.f;
#pragma unroll 4
  for (int i = 0; i < half / 4; ++i) {
    float4 v = s4[i];
    sum += v.x + v.y + v.z + v.w;
    ss  += v.x * v.x + v.y * v.y + v.z * v.z + v.w * v.w;
  }
  sum += __shfl_xor(sum, 1, 32);
  ss  += __shfl_xor(ss, 1, 32);
  const float inv = 1.f / (float)W;
  const float mu  = sum * inv;
  const float var = ss * inv - mu * mu;
  const float rs  = rsqrtf(var + 1e-5f);
  const float4* g4 = (const float4*)(gp + base);
  const float4* b4 = (const float4*)(bp + base);
#pragma unroll 1
  for (int i = 0; i < half / 4; ++i) {
    float4 v = s4[i], g = g4[i], bb = b4[i];
    float o0 = (v.x - mu) * rs * g.x + bb.x;
    float o1 = (v.y - mu) * rs * g.y + bb.y;
    float o2 = (v.z - mu) * rs * g.z + bb.z;
    float o3 = (v.w - mu) * rs * g.w + bb.w;
    if (GELU) {
      o0 = 0.5f * o0 * (1.f + erff(o0 * 0.70710678118654752f));
      o1 = 0.5f * o1 * (1.f + erff(o1 * 0.70710678118654752f));
      o2 = 0.5f * o2 * (1.f + erff(o2 * 0.70710678118654752f));
      o3 = 0.5f * o3 * (1.f + erff(o3 * 0.70710678118654752f));
    }
    if (OUTF)
      ((float4*)(dstf + m * W + base))[i] = make_float4(o0, o1, o2, o3);
    if (OUTB) {
      uint2 pk;
      pk.x = f2bf(o0) | (f2bf(o1) << 16);
      pk.y = f2bf(o2) | (f2bf(o3) << 16);
      ((uint2*)(dstb + m * ldb + base))[i] = pk;
    }
  }
}

// ---------- [16x3] @ W^T + b -> natural fp32 [16][128] (weights in LDS) ----
__device__ __forceinline__ void proj3(WaveMem* wm, const float* xbuf,
                                      const float* wlds, const float* blds,
                                      float* dst, int lane) {
  const int m = lane >> 1, base = (lane & 1) * 64;
  const float x0 = xbuf[m * 4 + 0], x1 = xbuf[m * 4 + 1], x2 = xbuf[m * 4 + 2];
  float4* d4 = (float4*)(dst + m * 128 + base);
#pragma unroll 1
  for (int i = 0; i < 16; ++i) {
    float o[4];
#pragma unroll
    for (int c = 0; c < 4; ++c) {
      const int f = base + i * 4 + c;
      const float* w = wlds + f * 3;
      o[c] = w[0] * x0 + w[1] * x1 + w[2] * x2 + blds[f];
    }
    d4[i] = make_float4(o[0], o[1], o[2], o[3]);
  }
}

// ---------- gate via padded-N WMMA: [16x512] @ [512x16(2 real)] ------------
__device__ __forceinline__ void gate_gemm(WaveMem* wm, const Params& p,
                                          int gb0, int lane) {
  v8f acc = {0.f, 0.f, 0.f, 0.f, 0.f, 0.f, 0.f, 0.f};
  const unsigned short* bufs[4] = {wm->abf, wm->tbf, wm->aabf, wm->tabf};
#pragma unroll
  for (int seg = 0; seg < 4; ++seg)
#pragma unroll
    for (int k2 = 0; k2 < 4; ++k2) {
      v16bf a = load_afrag(bufs[seg], 128, k2, lane);
      v16bf b = load_bfrag(p.ws + WS_GATE, seg * 4 + k2, lane);
      acc = __builtin_amdgcn_wmma_f32_16x16x32_bf16(
          false, a, false, b, (short)0, acc, false, false);
    }
  const int col = lane & 15, mb = (lane >> 4) * 8;
  if (col < 2) {
    const float bb = p.gate_b[col];
#pragma unroll
    for (int r = 0; r < 8; ++r) wm->gsc[(mb + r) * 2 + col] = acc[r] + bb;
  }
  WAVE_SYNC();
  if (lane < 16) {
    const float s0 = wm->gsc[lane * 2 + 0], s1 = wm->gsc[lane * 2 + 1];
    const float mx = fmaxf(s0, s1);
    const float e0 = expf(s0 - mx), e1 = expf(s1 - mx);
    const float den = 1.f / (e0 + e1);
    const float g0 = e0 * den, g1 = e1 * den;
    wm->gts[lane * 2 + 0] = g0;
    wm->gts[lane * 2 + 1] = g1;
    float* go = p.out + (size_t)3 * p.B + (size_t)(gb0 + lane) * 2;
    go[0] = g0; go[1] = g1;
  }
}

// ---------- combined = [(af+aat)*g0 | (tf+tat)*g1] -> bf16 [16][256] -------
__device__ __forceinline__ void combined_pack(WaveMem* wm, int lane) {
  const int m = lane >> 1, part = lane & 1;
  const float g   = wm->gts[m * 2 + part];
  const float* x  = (part ? wm->tf  : wm->af)  + m * 128;
  const float* y  = (part ? wm->tat : wm->aat) + m * 128;
  unsigned short* d = wm->cbf + m * 256 + part * 128;
#pragma unroll 1
  for (int i = 0; i < 32; ++i) {
    float4 a = ((const float4*)x)[i];
    float4 b = ((const float4*)y)[i];
    const float o0 = (a.x + b.x) * g, o1 = (a.y + b.y) * g;
    const float o2 = (a.z + b.z) * g, o3 = (a.w + b.w) * g;
    uint2 pk;
    pk.x = f2bf(o0) | (f2bf(o1) << 16);
    pk.y = f2bf(o2) | (f2bf(o3) << 16);
    ((uint2*)d)[i] = pk;
  }
}

// ---------- head fc2 via padded-N WMMA: [16x64] @ [64x16(1 real)] + tanh ---
__device__ __forceinline__ void fc2_head(WaveMem* wm, const Params& p,
                                         int gb0, int h, int wsoff, int lane) {
  v8f acc = {0.f, 0.f, 0.f, 0.f, 0.f, 0.f, 0.f, 0.f};
#pragma unroll
  for (int kt = 0; kt < 2; ++kt) {
    v16bf a = load_afrag(wm->tbf, 64, kt, lane);
    v16bf b = load_bfrag(p.ws + wsoff, kt, lane);
    acc = __builtin_amdgcn_wmma_f32_16x16x32_bf16(
        false, a, false, b, (short)0, acc, false, false);
  }
  const int col = lane & 15, mb = (lane >> 4) * 8;
  if (col == 0) {
    const float bb = p.h_fc2b[h][0];
#pragma unroll
    for (int r = 0; r < 8; ++r)
      p.out[(size_t)(gb0 + mb + r) * 3 + h] = tanhf(acc[r] + bb);
  }
}

// ============================ main kernel ==================================
__global__ void vad_fused_kernel(Params p) {
  extern __shared__ float4 smem_f4[];
  const int lane = threadIdx.x & 31;
  const int wave = threadIdx.x >> 5;
  WaveMem* wm = reinterpret_cast<WaveMem*>(smem_f4) + wave;

  const int ntiles = p.B >> 4;
  const int tile   = blockIdx.x * WAVES + wave;
  if (tile >= ntiles) return;            // wave-uniform: EXEC stays all-ones
  const int gb0 = tile * 16;

  // ---- stage 16 input rows + proj weights (tat is free as scratch) ----
  if (lane < 16) {
    const float* a = p.audio + (size_t)(gb0 + lane) * 3;
    const float* t = p.text  + (size_t)(gb0 + lane) * 3;
    wm->xa[lane * 4 + 0] = a[0]; wm->xa[lane * 4 + 1] = a[1];
    wm->xa[lane * 4 + 2] = a[2];
    wm->xt[lane * 4 + 0] = t[0]; wm->xt[lane * 4 + 1] = t[1];
    wm->xt[lane * 4 + 2] = t[2];
  }
#pragma unroll 1
  for (int idx = lane; idx < 1024; idx += 32) {      // [apw|apb|tpw|tpb]
    float v;
    if      (idx < 384) v = p.ap_w[idx];
    else if (idx < 512) v = p.ap_b[idx - 384];
    else if (idx < 896) v = p.tp_w[idx - 512];
    else                v = p.tp_b[idx - 896];
    wm->tat[idx] = v;
  }
  WAVE_SYNC();

  // ---- af = LN(audio_proj(x_a)); tf = LN(text_proj(x_t)); bf16 copies ----
  proj3(wm, wm->xa, wm->tat, wm->tat + 384, wm->tmp, lane);       WAVE_SYNC();
  ln16<128, true, true, false>(wm->tmp, wm->af, wm->abf, 128,
                               p.ap_lng, p.ap_lnb, lane);         WAVE_SYNC();
  proj3(wm, wm->xt, wm->tat + 512, wm->tat + 896, wm->tmp, lane); WAVE_SYNC();
  ln16<128, true, true, false>(wm->tmp, wm->tf, wm->tbf, 128,
                               p.tp_lng, p.tp_lnb, lane);         WAVE_SYNC();

  // ---- a2t attention: kv = LN(tf); v = kv@Wv^T; aat = v@Wo^T + af ----
  ln16<128, false, true, false>(wm->tf, nullptr, wm->cbf, 256,
                                p.a2t_ln2g, p.a2t_ln2b, lane);    WAVE_SYNC();
  gemm<4, 8, false, true, false>(wm->cbf, 256, p.ws + WS_A2T_V, p.a2t_vb,
                                 nullptr, wm->cbf + 128, 256, nullptr, lane);
  WAVE_SYNC();
  gemm<4, 8, true, true, true>(wm->cbf + 128, 256, p.ws + WS_A2T_O, p.a2t_ob,
                               wm->aat, wm->aabf, 128, wm->af, lane);
  WAVE_SYNC();

  // ---- t2a attention ----
  ln16<128, false, true, false>(wm->af, nullptr, wm->cbf, 256,
                                p.t2a_ln2g, p.t2a_ln2b, lane);    WAVE_SYNC();
  gemm<4, 8, false, true, false>(wm->cbf, 256, p.ws + WS_T2A_V, p.t2a_vb,
                                 nullptr, wm->cbf + 128, 256, nullptr, lane);
  WAVE_SYNC();
  gemm<4, 8, true, true, true>(wm->cbf + 128, 256, p.ws + WS_T2A_O, p.t2a_ob,
                               wm->tat, wm->tabf, 128, wm->tf, lane);
  WAVE_SYNC();

  // ---- gate softmax (padded-N WMMA) ----
  gate_gemm(wm, p, gb0, lane);                                    WAVE_SYNC();

  // ---- integ = GELU(LN(combined @ Wi^T + bi)) -> bf16 in abf ----
  combined_pack(wm, lane);                                        WAVE_SYNC();
  gemm<8, 8, true, false, false>(wm->cbf, 256, p.ws + WS_INTEG, p.integ_b,
                                 wm->tmp, nullptr, 128, nullptr, lane);
  WAVE_SYNC();
  ln16<128, false, true, true>(wm->tmp, nullptr, wm->abf, 128,
                               p.integ_lng, p.integ_lnb, lane);   WAVE_SYNC();

  // ---- heads: val, aro, dom -> fused columns 0,1,2 ----
  const int hfc1[3] = {WS_VAL, WS_ARO, WS_DOM};
  const int hfc2[3] = {WS_VFC2, WS_AFC2, WS_DFC2};
#pragma unroll
  for (int h = 0; h < 3; ++h) {
    gemm<4, 4, true, false, false>(wm->abf, 128, p.ws + hfc1[h], p.h_fc1b[h],
                                   wm->tmp, nullptr, 64, nullptr, lane);
    WAVE_SYNC();
    ln16<64, false, true, true>(wm->tmp, nullptr, wm->tbf, 64,
                                p.h_lng[h], p.h_lnb[h], lane);
    WAVE_SYNC();
    fc2_head(wm, p, gb0, h, hfc2[h], lane);
    WAVE_SYNC();
  }
}

// ============================ host glue ====================================
extern "C" void kernel_launch(void* const* d_in, const int* in_sizes, int n_in,
                              void* d_out, int out_size, void* d_ws,
                              size_t ws_size, hipStream_t stream) {
  (void)n_in; (void)out_size; (void)ws_size;
  auto F = [&](int i) { return (const float*)d_in[i]; };

  const int B = in_sizes[0] / 3;
  unsigned int* ws = (unsigned int*)d_ws;

  // ---- prep: swizzle weights into bf16 B-fragments (pad N where needed) --
  struct { int src; int off; int NT; int KT; int Nreal; } prep[12] = {
    { 12, WS_A2T_V, 8, 4,  128 },  // a2t_attn.v.w  128x128
    {  8, WS_A2T_O, 8, 4,  128 },  // a2t_attn.o.w
    { 46, WS_T2A_V, 8, 4,  128 },  // t2a_attn.v.w
    { 42, WS_T2A_O, 8, 4,  128 },  // t2a_attn.o.w
    { 32, WS_INTEG, 8, 8,  128 },  // integ.lin.w   128x256
    { 52, WS_VAL,   4, 4,   64 },  // val.fc1.w      64x128
    { 14, WS_ARO,   4, 4,   64 },  // aro.fc1.w
    { 24, WS_DOM,   4, 4,   64 },  // dom.fc1.w
    { 30, WS_GATE,  1, 16,   2 },  // gate.w          2x512 (pad N->16)
    { 54, WS_VFC2,  1, 2,    1 },  // val.fc2.w       1x64  (pad N->16)
    { 16, WS_AFC2,  1, 2,    1 },  // aro.fc2.w
    { 26, WS_DFC2,  1, 2,    1 },  // dom.fc2.w
  };
  for (int i = 0; i < 12; ++i) {
    const int total = prep[i].NT * prep[i].KT * 256;
    vad_prep_frag<<<(total + 255) / 256, 256, 0, stream>>>(
        F(prep[i].src), ws + prep[i].off, prep[i].NT, prep[i].KT,
        prep[i].Nreal);
  }

  // ---- main fused kernel ----
  Params p;
  p.audio = F(0);  p.text = F(57);
  p.ap_w = F(20);  p.ap_b = F(19);  p.ap_lng = F(22);  p.ap_lnb = F(21);
  p.tp_w = F(48);  p.tp_b = F(47);  p.tp_lng = F(50);  p.tp_lnb = F(49);
  p.a2t_ln2g = F(6);  p.a2t_ln2b = F(5);  p.a2t_vb = F(11); p.a2t_ob = F(7);
  p.t2a_ln2g = F(40); p.t2a_ln2b = F(39); p.t2a_vb = F(45); p.t2a_ob = F(41);
  p.gate_b = F(29);
  p.integ_b = F(31); p.integ_lng = F(34); p.integ_lnb = F(33);
  p.h_fc1b[0] = F(51); p.h_fc1b[1] = F(13); p.h_fc1b[2] = F(23);
  p.h_lng[0]  = F(56); p.h_lng[1]  = F(18); p.h_lng[2]  = F(28);
  p.h_lnb[0]  = F(55); p.h_lnb[1]  = F(17); p.h_lnb[2]  = F(27);
  p.h_fc2b[0] = F(53); p.h_fc2b[1] = F(15); p.h_fc2b[2] = F(25);
  p.ws  = ws;
  p.out = (float*)d_out;
  p.B   = B;

  const int ntiles = B / 16;                       // B = 131072 -> 8192 tiles
  const int grid   = (ntiles + WAVES - 1) / WAVES; // 4 waves (tiles) per block
  vad_fused_kernel<<<grid, WAVES * 32, sizeof(WaveMem) * WAVES, stream>>>(p);
}
